// RNNModel_16509854285990
// MI455X (gfx1250) — matrix operand order repaired and checked by reading, not compiled, not run on hardware
//
#include <hip/hip_runtime.h>
#include <hip/hip_bf16.h>
#include <stdint.h>

typedef __attribute__((ext_vector_type(16))) _Float16 v16h;
typedef __attribute__((ext_vector_type(8)))  _Float16 v8h;
typedef __attribute__((ext_vector_type(8)))  float    v8f;

typedef __attribute__((ext_vector_type(4))) unsigned int u32x4;
typedef __attribute__((ext_vector_type(8))) int          i32x8;
typedef __attribute__((ext_vector_type(4))) int          i32x4;

#ifndef __has_builtin
#define __has_builtin(x) 0
#endif
#if __has_builtin(__builtin_amdgcn_tensor_load_to_lds) && \
    __has_builtin(__builtin_amdgcn_s_wait_tensorcnt)
#define USE_TDM 1
#else
#define USE_TDM 0
#endif

// ---------------------------------------------------------------------------
// Fragment helpers (WMMA 16x16x32 f16 operand k order used here, wave32)
//   A (16x32, M x K): lane L holds row M=L%16; g=L/16:
//     halves j=0..15 -> K = 16*g + j
//   B (32x16, K x N): lane L holds col N=L%16; halves j=0..15 -> K = 16*g + j
//   C/D (16x16 f32): VGPR i, lane L -> row M = 8*g + i, col N = L%16
// ---------------------------------------------------------------------------
__device__ __forceinline__ v16h combine88(v8h lo, v8h hi) {
  v16h r;
#pragma unroll
  for (int j = 0; j < 8; ++j) { r[j] = lo[j]; r[j + 8] = hi[j]; }
  return r;
}

__device__ __forceinline__ v16h make_a_frag(const _Float16* __restrict__ rowp, int g) {
  v8h lo = *reinterpret_cast<const v8h*>(rowp + g * 16);
  v8h hi = *reinterpret_cast<const v8h*>(rowp + g * 16 + 8);
  return combine88(lo, hi);
}

__device__ __forceinline__ v8h cvt8(const float* __restrict__ p) {
  float4 f0 = *reinterpret_cast<const float4*>(p);
  float4 f1 = *reinterpret_cast<const float4*>(p + 4);
  v8h v;
  v[0] = (_Float16)f0.x; v[1] = (_Float16)f0.y; v[2] = (_Float16)f0.z; v[3] = (_Float16)f0.w;
  v[4] = (_Float16)f1.x; v[5] = (_Float16)f1.y; v[6] = (_Float16)f1.z; v[7] = (_Float16)f1.w;
  return v;
}

__device__ __forceinline__ v16h cvt16(const float* __restrict__ p) {
  return combine88(cvt8(p), cvt8(p + 8));
}

// A fragment built from an f32-staged LDS row
__device__ __forceinline__ v16h a_frag_f32(const float* __restrict__ rowp, int g) {
  return combine88(cvt8(rowp + g * 16), cvt8(rowp + g * 16 + 8));
}

// ---------------------------------------------------------------------------
// Tensor Data Mover: 2D tile (tile_w x tile_h elements, row stride in elems)
// from global to LDS.  D# built per CDNA5 ISA 8.3/8.4; issued by one wave.
//   ds_code: 0=1B, 1=2B, 2=4B elements
// ---------------------------------------------------------------------------
#if USE_TDM
__device__ __forceinline__ void tdm_load_2d(unsigned lds_off,
                                            unsigned long long gaddr,
                                            unsigned ds_code,
                                            unsigned tile_w, unsigned tile_h,
                                            unsigned long long row_stride) {
  u32x4 g0;
  g0[0] = 1u;                                            // count=1, user desc
  g0[1] = lds_off;                                       // lds_addr (bytes)
  g0[2] = (unsigned)(gaddr & 0xffffffffull);             // global_addr[31:0]
  g0[3] = (unsigned)((gaddr >> 32) & 0x1ffffffull)       // global_addr[56:32]
          | (2u << 30);                                  // type = 2 (image)
  unsigned td0 = tile_w, td1 = tile_h;                   // tensor dims = tile (no OOB)
  i32x8 g1;
  g1[0] = (int)(ds_code << 16);                          // mask=0, data_size
  g1[1] = (int)((td0 & 0xffffu) << 16);                  // tensor_dim0 lo
  g1[2] = (int)((td0 >> 16) | ((td1 & 0xffffu) << 16));  // dim0 hi | dim1 lo
  g1[3] = (int)((td1 >> 16) | (tile_w << 16));           // dim1 hi | tile_dim0
  g1[4] = (int)(tile_h & 0xffffu);                       // tile_dim1 (tile_dim2=0)
  g1[5] = (int)(unsigned)(row_stride & 0xffffffffull);   // tensor_dim0_stride lo
  g1[6] = (int)(unsigned)(row_stride >> 32);             // stride hi (dim1_stride=0)
  g1[7] = 0;
  i32x4 z4 = (i32x4)0;
#if defined(__clang_major__) && (__clang_major__ >= 23)
  i32x8 z8 = (i32x8)0;
  __builtin_amdgcn_tensor_load_to_lds(g0, g1, z4, z4, z8, 0);
#else
  __builtin_amdgcn_tensor_load_to_lds(g0, g1, z4, z4, 0);
#endif
}
#endif

// ---------------------------------------------------------------------------
// Kernel 1: Out[M,N] = A[M,K] * W[N,K]^T + (b0[n]+b1[n])   (f32 out)
// Block tile 64(M) x 128(N), 256 threads = 8 waves; wave w owns N-subtile w,
// all 4 M-subtiles.  K staged in 32-wide chunks, TDM double-buffered: chunk
// kc+1 DMAs into LDS while chunk kc computes (s_wait_tensorcnt(2) retires the
// older A/B pair since TDM completes in order within a wave).
// ---------------------------------------------------------------------------
template <bool A_HALF>
__global__ void __launch_bounds__(256)
wmma_gemm_bias(const void* __restrict__ Aop, int lda,
               const float* __restrict__ W, int K, int N,
               const float* __restrict__ b0, const float* __restrict__ b1,
               float* __restrict__ Out) {
  constexpr int EASZ = A_HALF ? 2 : 4;                 // A element bytes
  __shared__ __align__(16) unsigned char AsRaw[2][64 * 32 * EASZ];
  __shared__ __align__(16) float         Bs[2][128 * 32];

  const int tid  = threadIdx.x;
  const int wave = tid >> 5;
  const int lane = tid & 31;
  const int g    = lane >> 4;
  const int ln16 = lane & 15;

  const long mbase = (long)blockIdx.x * 64;
  const int  nbase = blockIdx.y * 128;
  const char* Abase = (const char*)Aop;

  v8f acc[4];
#pragma unroll
  for (int i = 0; i < 4; ++i) acc[i] = (v8f){};

  const int nchunks = K >> 5;

#if USE_TDM
  // prologue: DMA chunk 0 into buffer 0 (wave 0 drives the TDM)
  if (tid < 32) {
    unsigned long long ga = (unsigned long long)(Abase) +
                            (unsigned long long)((mbase * (long)lda) * EASZ);
    tdm_load_2d((unsigned)(size_t)&AsRaw[0][0], ga, A_HALF ? 1u : 2u,
                32u, 64u, (unsigned long long)lda);
    unsigned long long gb = (unsigned long long)(const char*)(W + (long)nbase * K);
    tdm_load_2d((unsigned)(size_t)&Bs[0][0], gb, 2u, 32u, 128u,
                (unsigned long long)K);
  }
#endif

  for (int kc = 0; kc < nchunks; ++kc) {
    const int cur = kc & 1;
#if USE_TDM
    if (kc + 1 < nchunks) {
      if (tid < 32) {
        unsigned long long ga = (unsigned long long)(Abase) +
            (unsigned long long)((mbase * (long)lda + (long)(kc + 1) * 32) * EASZ);
        tdm_load_2d((unsigned)(size_t)&AsRaw[cur ^ 1][0], ga, A_HALF ? 1u : 2u,
                    32u, 64u, (unsigned long long)lda);
        unsigned long long gb = (unsigned long long)
            (const char*)(W + (long)nbase * K + (long)(kc + 1) * 32);
        tdm_load_2d((unsigned)(size_t)&Bs[cur ^ 1][0], gb, 2u, 32u, 128u,
                    (unsigned long long)K);
        __builtin_amdgcn_s_wait_tensorcnt(2);  // pair for chunk kc retired
      }
    } else {
      if (tid < 32) __builtin_amdgcn_s_wait_tensorcnt(0);
    }
    __syncthreads();
#else
    // synchronous fallback staging
    {
      int row = tid >> 2;
      int c   = (tid & 3) * 8;
      long src = (mbase + row) * (long)lda + (long)kc * 32 + c;
      if constexpr (A_HALF) {
        const _Float16* Ah = (const _Float16*)Aop;
        *reinterpret_cast<v8h*>(&AsRaw[cur][(row * 32 + c) * 2]) =
            *reinterpret_cast<const v8h*>(Ah + src);
      } else {
        const float* Af = (const float*)Aop;
        float* dst = (float*)&AsRaw[cur][(row * 32 + c) * 4];
        *reinterpret_cast<float4*>(dst)     = *reinterpret_cast<const float4*>(Af + src);
        *reinterpret_cast<float4*>(dst + 4) = *reinterpret_cast<const float4*>(Af + src + 4);
      }
      int brow = tid >> 1;
      int bc   = (tid & 1) * 16;
      long bsrc = (long)(nbase + brow) * K + (long)kc * 32 + bc;
#pragma unroll
      for (int j = 0; j < 16; j += 4)
        *reinterpret_cast<float4*>(&Bs[cur][brow * 32 + bc + j]) =
            *reinterpret_cast<const float4*>(W + bsrc + j);
    }
    __syncthreads();
#endif

    v16h bf = cvt16(&Bs[cur][(wave * 16 + ln16) * 32 + g * 16]);
#pragma unroll
    for (int ms = 0; ms < 4; ++ms) {
      v16h af;
      if constexpr (A_HALF) {
        af = make_a_frag((const _Float16*)&AsRaw[cur][(ms * 16 + ln16) * 32 * 2], g);
      } else {
        af = a_frag_f32((const float*)&AsRaw[cur][(ms * 16 + ln16) * 32 * 4], g);
      }
      acc[ms] = __builtin_amdgcn_wmma_f32_16x16x32_f16(
          false, af, false, bf, (short)0, acc[ms], false, false);
    }
    __syncthreads();  // buffer `cur` free before it is re-filled at kc+2
  }

  const int col = nbase + wave * 16 + ln16;
  const float bias = b0[col] + b1[col];
#pragma unroll
  for (int ms = 0; ms < 4; ++ms) {
#pragma unroll
    for (int i = 0; i < 8; ++i) {
      long rowg = mbase + ms * 16 + g * 8 + i;
      Out[rowg * N + col] = acc[ms][i] + bias;
    }
  }
}

// ---------------------------------------------------------------------------
// Device-wide barrier (32 co-resident blocks), monotonic counter + s_sleep.
// ---------------------------------------------------------------------------
__device__ __forceinline__ void global_barrier(unsigned int* bar, unsigned int target) {
  __threadfence();
  __syncthreads();
  if (threadIdx.x == 0) {
    atomicAdd(bar, 1u);
    volatile unsigned int* vb = (volatile unsigned int*)bar;
    while (*vb < target) { __builtin_amdgcn_s_sleep(1); }
  }
  __syncthreads();
  __threadfence();
}

// ---------------------------------------------------------------------------
// Kernel 2: recurrence  h <- tanh(xp_t + h * W_hh^T), T sequential steps.
// 32 blocks x 128 threads (4 waves). Block j owns output cols [16j,16j+16):
// its 16x512 W_hh slice is TDM-DMA'd to LDS (contiguous 32 KB), converted to
// f16 once, then reused for all 2048 steps.  Wave w computes M-tile w with 16
// chained WMMAs over K=512.  h ping-pongs between two f16 L2-resident
// buffers; one device barrier per step.
// ---------------------------------------------------------------------------
template <bool STORE_SEQ>
__global__ void __launch_bounds__(128)
rnn_scan(const float* __restrict__ xp,     // [B*T, 512] f32
         const float* __restrict__ Whh,    // [512, 512] f32
         const float* __restrict__ h0,     // [512] f32
         _Float16* __restrict__ hseq,      // [B*T, 512] f16 (layer-0 outputs)
         _Float16* __restrict__ hbuf,      // [2][64*512] f16 ping-pong
         unsigned int* __restrict__ bar,
         int T) {
  __shared__ __align__(32) _Float16 Ws[16 * 512];   // 16 KB f16 slice
#if USE_TDM
  __shared__ __align__(16) float Wraw[16 * 512];    // 32 KB f32 staging
#endif

  const int tid  = threadIdx.x;
  const int wave = tid >> 5;   // M-tile
  const int lane = tid & 31;
  const int g    = lane >> 4;
  const int ln16 = lane & 15;
  const int n0   = blockIdx.x * 16;

  // ---- bring this block's W_hh slice into LDS as f16 ----
#if USE_TDM
  if (tid < 32) {
    // rows n0..n0+15 of row-major [512,512] are one contiguous 8192-elem run
    tdm_load_2d((unsigned)(size_t)&Wraw[0],
                (unsigned long long)(const char*)(Whh + (long)n0 * 512),
                2u, 8192u, 1u, 8192ull);
    __builtin_amdgcn_s_wait_tensorcnt(0);
  }
  __syncthreads();
  {
    int wrow = tid >> 3;          // 0..15
    int wcol = (tid & 7) * 64;    // 0..448
    const float* wsrc = &Wraw[wrow * 512 + wcol];
    _Float16* wdst = &Ws[wrow * 512 + wcol];
#pragma unroll
    for (int j = 0; j < 64; j += 8)
      *reinterpret_cast<v8h*>(wdst + j) = cvt8(wsrc + j);
  }
#else
  {
    int wrow = tid >> 3;
    int wcol = (tid & 7) * 64;
    const float* wsrc = Whh + (long)(n0 + wrow) * 512 + wcol;
    _Float16* wdst = &Ws[wrow * 512 + wcol];
#pragma unroll
    for (int j = 0; j < 64; j += 8)
      *reinterpret_cast<v8h*>(wdst + j) = cvt8(wsrc + j);
  }
#endif
  __syncthreads();

  // ---- init h (broadcast h0 over batch) into buffer 0, our 16 columns ----
  for (int idx = tid; idx < 64 * 16; idx += 128) {
    int r = idx >> 4, c = idx & 15;
    hbuf[r * 512 + n0 + c] = (_Float16)h0[n0 + c];
  }

  const unsigned int nb = gridDim.x;
  unsigned int target = nb;
  global_barrier(bar, target);

  int p = 0;
  for (int t = 0; t < T; ++t) {
    const _Float16* hin  = hbuf + p * (64 * 512);
    _Float16*       hout = hbuf + (p ^ 1) * (64 * 512);

    v8f acc = (v8f){};
    const _Float16* arow = hin + (long)(wave * 16 + ln16) * 512;
    const _Float16* brow = &Ws[ln16 * 512];
#pragma unroll
    for (int kk = 0; kk < 16; ++kk) {
      v16h af = make_a_frag(arow + kk * 32, g);
      v16h bf = *reinterpret_cast<const v16h*>(brow + kk * 32 + g * 16);
      acc = __builtin_amdgcn_wmma_f32_16x16x32_f16(
          false, af, false, bf, (short)0, acc, false, false);
    }

    const int col = n0 + ln16;
    if (t + 1 < T)  // warm next step's xp slice (global_prefetch_b8)
      __builtin_prefetch(&xp[((long)(wave * 16 + ln16) * T + (t + 1)) * 512 + col], 0, 1);
#pragma unroll
    for (int i = 0; i < 8; ++i) {
      int r = wave * 16 + g * 8 + i;               // batch row
      float v = tanhf(acc[i] + xp[((long)r * T + t) * 512 + col]);
      _Float16 hv = (_Float16)v;
      hout[(long)r * 512 + col] = hv;
      if constexpr (STORE_SEQ) hseq[((long)r * T + t) * 512 + col] = hv;
    }

    target += nb;
    global_barrier(bar, target);
    p ^= 1;
  }
}

// ---------------------------------------------------------------------------
// Kernel 3: out[64,128] = hT[64,512](f16) * W_fc[128,512]^T + b_fc  (f32)
// One block, 8 waves: wave w -> N-subtile w, 4 M-tiles, K=512.
// ---------------------------------------------------------------------------
__global__ void __launch_bounds__(256)
head_gemm(const _Float16* __restrict__ h,   // [64,512] f16
          const float* __restrict__ Wfc,    // [128,512] f32
          const float* __restrict__ bfc,    // [128]
          float* __restrict__ out) {        // [64,128]
  const int tid  = threadIdx.x;
  const int wave = tid >> 5;
  const int lane = tid & 31;
  const int g    = lane >> 4;
  const int ln16 = lane & 15;
  const int n0   = wave * 16;

  v8f acc[4];
#pragma unroll
  for (int i = 0; i < 4; ++i) acc[i] = (v8f){};

  for (int kk = 0; kk < 16; ++kk) {
    v16h bf = cvt16(Wfc + (long)(n0 + ln16) * 512 + kk * 32 + g * 16);
#pragma unroll
    for (int ms = 0; ms < 4; ++ms) {
      v16h af = make_a_frag(h + (long)(ms * 16 + ln16) * 512 + kk * 32, g);
      acc[ms] = __builtin_amdgcn_wmma_f32_16x16x32_f16(
          false, af, false, bf, (short)0, acc[ms], false, false);
    }
  }

  const int col = n0 + ln16;
  const float bias = bfc[col];
#pragma unroll
  for (int ms = 0; ms < 4; ++ms)
#pragma unroll
    for (int i = 0; i < 8; ++i)
      out[(ms * 16 + g * 8 + i) * 128 + col] = acc[ms][i] + bias;
}

// ---------------------------------------------------------------------------
extern "C" void kernel_launch(void* const* d_in, const int* in_sizes, int n_in,
                              void* d_out, int out_size, void* d_ws, size_t ws_size,
                              hipStream_t stream) {
  (void)in_sizes; (void)n_in; (void)out_size; (void)ws_size;

  const float* x    = (const float*)d_in[0];
  const float* Wih0 = (const float*)d_in[1];
  const float* Whh0 = (const float*)d_in[2];
  const float* bih0 = (const float*)d_in[3];
  const float* bhh0 = (const float*)d_in[4];
  const float* h00  = (const float*)d_in[5];
  const float* Wih1 = (const float*)d_in[6];
  const float* Whh1 = (const float*)d_in[7];
  const float* bih1 = (const float*)d_in[8];
  const float* bhh1 = (const float*)d_in[9];
  const float* h01  = (const float*)d_in[10];
  const float* Wfc  = (const float*)d_in[11];
  const float* bfc  = (const float*)d_in[12];
  float* out = (float*)d_out;

  const int Bb = 64, T = 2048, F = 256, H = 512;
  const long Mtot = (long)Bb * T;

  char* ws = (char*)d_ws;
  size_t off = 0;
  float*     xp   = (float*)(ws + off);     off += (size_t)Mtot * H * sizeof(float);   // 256 MB, reused
  _Float16*  hseq = (_Float16*)(ws + off);  off += (size_t)Mtot * H * sizeof(_Float16);// 128 MB
  _Float16*  hbuf = (_Float16*)(ws + off);  off += (size_t)2 * 64 * 512 * sizeof(_Float16);
  off = (off + 255) & ~(size_t)255;
  unsigned int* bar = (unsigned int*)(ws + off);

  dim3 gg(Mtot / 64, H / 128), gb(256);

  // Layer 0: input projection + scan (stores full sequence as f16)
  hipMemsetAsync(bar, 0, sizeof(unsigned int), stream);
  wmma_gemm_bias<false><<<gg, gb, 0, stream>>>(x, F, Wih0, F, H, bih0, bhh0, xp);
  rnn_scan<true><<<dim3(32), dim3(128), 0, stream>>>(xp, Whh0, h00, hseq, hbuf, bar, T);

  // Layer 1: input projection (f16 A directly) + scan (final state only)
  hipMemsetAsync(bar, 0, sizeof(unsigned int), stream);
  wmma_gemm_bias<true><<<gg, gb, 0, stream>>>(hseq, H, Wih1, H, H, bih1, bhh1, xp);
  rnn_scan<false><<<dim3(32), dim3(128), 0, stream>>>(xp, Whh1, h01, hseq, hbuf, bar, T);

  // Head: T=2048 is even, so the final hidden state lands in hbuf[0].
  head_gemm<<<dim3(1), dim3(256), 0, stream>>>(hbuf, Wfc, bfc, out);
}
